// BertSelfAttention_377957122362
// MI455X (gfx1250) — matrix operand order repaired and checked by reading, compiled, not run on hardware
//
#include <hip/hip_runtime.h>
#include <hip/hip_bf16.h>
#include <stdint.h>

// Problem constants (from reference)
#define B_   4
#define S_   2048
#define H_   768
#define NH_  12
#define HD_  64
#define MT_  (B_ * S_)   // 8192 token rows

#define KSTR 72          // padded LDS row stride (elements) -> bank-conflict-free b128

typedef __attribute__((ext_vector_type(16))) __bf16 v16bf;
typedef __attribute__((ext_vector_type(8)))  float  v8f;
typedef unsigned short us;

// ---------------------------------------------------------------- helpers
__device__ __forceinline__ us f2bf(float f) {
    uint32_t u = __float_as_uint(f);
    u += 0x7FFFu + ((u >> 16) & 1u);     // round-to-nearest-even
    return (us)(u >> 16);
}

__device__ __forceinline__ v16bf frag_from(uint4 lo, uint4 hi) {
    union { v16bf v; uint4 u[2]; } x;
    x.u[0] = lo; x.u[1] = hi;
    return x.v;
}

__device__ __forceinline__ v8f wmma_bf16(v16bf a, v16bf b, v8f c) {
    return __builtin_amdgcn_wmma_f32_16x16x32_bf16(
        /*neg_a=*/false, a, /*neg_b=*/false, b,
        /*c_mod=*/(short)0, c, /*reuse_a=*/false, /*reuse_b=*/false);
}

// CDNA5 async global->LDS copy (GVS form): 16B per lane, tracked by ASYNCcnt.
// lds_off: LDS byte offset (vdst); goff: per-lane byte offset; base: uniform.
__device__ __forceinline__ void async_b128(unsigned lds_off, unsigned goff,
                                           unsigned long long base) {
    asm volatile("global_load_async_to_lds_b128 %0, %1, %2"
                 :: "v"(lds_off), "v"(goff), "s"(base) : "memory");
}
__device__ __forceinline__ void wait_async() {
    asm volatile("s_wait_asynccnt 0x0" ::: "memory");
}
__device__ __forceinline__ unsigned lds_addr(const void* p) {
    return (unsigned)(uintptr_t)p;       // low 32 bits of shared aperture = LDS offset
}

// ---------------------------------------------------------------- stage 1
// f32 -> bf16 conversion, 8 elements / thread, vectorized.
__global__ void cvt_bf16(const float* __restrict__ src, us* __restrict__ dst, int n) {
    int i = (blockIdx.x * blockDim.x + threadIdx.x) * 8;
    if (i >= n) return;
    const float4* s = reinterpret_cast<const float4*>(src + i);
    float4 a = s[0], b = s[1];
    union { uint4 u; us h[8]; } o;
    o.h[0] = f2bf(a.x); o.h[1] = f2bf(a.y); o.h[2] = f2bf(a.z); o.h[3] = f2bf(a.w);
    o.h[4] = f2bf(b.x); o.h[5] = f2bf(b.y); o.h[6] = f2bf(b.z); o.h[7] = f2bf(b.w);
    *reinterpret_cast<uint4*>(dst + i) = o.u;
}

// ---------------------------------------------------------------- stage 2
// Y = X @ W^T + b for W in {Wq,Wk,Wv}; one wave computes a 64x16 tile.
// Software-pipelined: next K-step's 5 fragments load while 4 WMMAs execute.
// Epilogue scatters: Q,K -> [b][h][s][d] (Q scaled 1/8), V -> [b][h][d][s].
__global__ void __launch_bounds__(256)
qkv_gemm(const us* __restrict__ X, const us* __restrict__ W3,
         const float* __restrict__ bq, const float* __restrict__ bk,
         const float* __restrict__ bv,
         us* __restrict__ Q, us* __restrict__ K, us* __restrict__ Vt) {
    const int wave = threadIdx.x >> 5;
    const int lane = threadIdx.x & 31;
    const int lo   = lane & 15;
    const int hi   = lane >> 4;

    // 128 mtiles(64 rows) * 48 ntiles * 3 matrices = 18432 wave tasks
    const int wg    = blockIdx.x * 8 + wave;
    const int mtile = wg / 144;
    const int rem   = wg % 144;
    const int ntile = rem / 3;
    const int mat   = rem % 3;

    const us* Wm = W3 + (size_t)mat * (H_ * H_);
    const int n  = ntile * 16 + lo;          // B col this lane carries

    const uint4* xr0 = reinterpret_cast<const uint4*>(X + (size_t)(mtile * 64 + lo     ) * H_);
    const uint4* xr1 = reinterpret_cast<const uint4*>(X + (size_t)(mtile * 64 + lo + 16) * H_);
    const uint4* xr2 = reinterpret_cast<const uint4*>(X + (size_t)(mtile * 64 + lo + 32) * H_);
    const uint4* xr3 = reinterpret_cast<const uint4*>(X + (size_t)(mtile * 64 + lo + 48) * H_);
    const uint4* wr  = reinterpret_cast<const uint4*>(Wm + (size_t)n * H_);

    // prologue fragments (K-step 0)
    v16bf bf  = frag_from(wr [hi],     wr [2 + hi]);       // B: k = ((e>>3)<<4)+hi*8+(e&7)
    v16bf f0  = frag_from(xr0[hi],     xr0[2 + hi]);       // A: k = ((e>>3)<<4)+hi*8+(e&7)
    v16bf f1  = frag_from(xr1[hi],     xr1[2 + hi]);
    v16bf f2  = frag_from(xr2[hi],     xr2[2 + hi]);
    v16bf f3  = frag_from(xr3[hi],     xr3[2 + hi]);

    v8f a0 = {}, a1 = {}, a2 = {}, a3 = {};
    for (int ks = 0; ks < H_ - 32; ks += 32) {
        const int kw = (ks + 32) >> 3;       // next K-step, uint4 units
        // issue next iteration's loads before consuming current fragments
        v16bf nb = frag_from(wr [kw + hi],     wr [kw + 2 + hi]);
        v16bf n0 = frag_from(xr0[kw + hi],     xr0[kw + 2 + hi]);
        v16bf n1 = frag_from(xr1[kw + hi],     xr1[kw + 2 + hi]);
        v16bf n2 = frag_from(xr2[kw + hi],     xr2[kw + 2 + hi]);
        v16bf n3 = frag_from(xr3[kw + hi],     xr3[kw + 2 + hi]);
        a0 = wmma_bf16(f0, bf, a0);
        a1 = wmma_bf16(f1, bf, a1);
        a2 = wmma_bf16(f2, bf, a2);
        a3 = wmma_bf16(f3, bf, a3);
        bf = nb; f0 = n0; f1 = n1; f2 = n2; f3 = n3;
    }
    // peeled last K-step
    a0 = wmma_bf16(f0, bf, a0);
    a1 = wmma_bf16(f1, bf, a1);
    a2 = wmma_bf16(f2, bf, a2);
    a3 = wmma_bf16(f3, bf, a3);

    const float* bias = (mat == 0) ? bq : (mat == 1) ? bk : bv;
    const float bval  = bias[n];
    const int   hIdx  = n >> 6;              // head
    const int   d     = n & 63;              // dim within head

    v8f accs[4] = {a0, a1, a2, a3};
    #pragma unroll
    for (int i = 0; i < 4; ++i) {
        #pragma unroll
        for (int r = 0; r < 8; ++r) {
            const int row  = mtile * 64 + i * 16 + hi * 8 + r;   // C layout: M = hi*8 + r
            const int bidx = row >> 11;                          // row / 2048
            const int s    = row & (S_ - 1);
            const float v  = accs[i][r] + bval;
            const size_t bh = (size_t)(bidx * NH_ + hIdx);
            if (mat == 0)      Q [(bh * S_  + s) * HD_ + d] = f2bf(v * 0.125f);
            else if (mat == 1) K [(bh * S_  + s) * HD_ + d] = f2bf(v);
            else               Vt[(bh * HD_ + d) * S_  + s] = f2bf(v);   // transposed
        }
    }
}

// ---------------------------------------------------------------- stage 3
// Flash attention. One block = one (b,h) and 128 q rows (8 waves x 16-row
// strips). K and V^T 64-key tiles staged in LDS via CDNA5 async global->LDS
// copies (double buffered, padded stride), shared by all 8 waves.
__global__ void __launch_bounds__(256)
flash_attn(const us* __restrict__ Q, const us* __restrict__ K, const us* __restrict__ Vt,
           const float* __restrict__ mask, float* __restrict__ out) {
    __shared__ us Ktile[2][64 * KSTR];     // [key][d]  padded rows
    __shared__ us Vtile[2][64 * KSTR];     // [d][key]  padded rows
    __shared__ us Pbuf[8][16 * KSTR];      // per-wave P staging, padded rows

    const int tid  = threadIdx.x;
    const int wave = tid >> 5;
    const int lane = tid & 31;
    const int lo   = lane & 15;
    const int hi   = lane >> 4;

    // block task: (b, h, qblock of 128 rows); B*NH*16 = 768 blocks
    const int blk  = blockIdx.x;
    const int qblk = blk & 15;
    const int h    = (blk >> 4) % NH_;
    const int b    = blk / (16 * NH_);
    const size_t bh = (size_t)(b * NH_ + h);
    const int qbase = qblk * 128 + wave * 16;

    // cooperative tile loader mapping: 256 threads, 64 rows x 128B, 32B/thread
    const int ldrow = tid >> 2;
    const int ldcol = (tid & 3) * 16;      // element offset within row

    // per-lane global byte offsets (within a tile) and LDS byte offsets
    const unsigned kgl = (unsigned)((ldrow * HD_ + ldcol) * sizeof(us));
    const unsigned vgl = (unsigned)((ldrow * S_  + ldcol) * sizeof(us));

    // Persistent Q A-fragments (16 rows x 64 dims -> 2 fragments)
    const uint4* qrow = reinterpret_cast<const uint4*>(Q + (bh * S_ + qbase + lo) * HD_);
    const v16bf qa0 = frag_from(qrow[hi],     qrow[2 + hi]);   // d 0..31
    const v16bf qa1 = frag_from(qrow[4 + hi], qrow[6 + hi]);   // d 32..63

    v8f o0 = {}, o1 = {}, o2 = {}, o3 = {};
    float m_run[8], l_run[8];
    #pragma unroll
    for (int r = 0; r < 8; ++r) { m_run[r] = -3.0e38f; l_run[r] = 0.f; }

    // prologue: async-stage tile 0 into buffer 0
    {
        const unsigned long long kbase = (unsigned long long)(const void*)(K  + bh * S_ * HD_);
        const unsigned long long vbase = (unsigned long long)(const void*)(Vt + bh * HD_ * S_);
        const unsigned klds = lds_addr(&Ktile[0][ldrow * KSTR + ldcol]);
        const unsigned vlds = lds_addr(&Vtile[0][ldrow * KSTR + ldcol]);
        async_b128(klds,      kgl,      kbase);
        async_b128(klds + 16, kgl + 16, kbase);
        async_b128(vlds,      vgl,      vbase);
        async_b128(vlds + 16, vgl + 16, vbase);
        wait_async();
    }
    __syncthreads();

    for (int kb = 0; kb < S_; kb += 64) {
        const int  buf = (kb >> 6) & 1;
        const bool pf  = (kb + 64) < S_;

        // kick off async copy of the next K/V tile into the other buffer;
        // it runs in the background for the whole compute section below.
        if (pf) {
            const unsigned long long kbase =
                (unsigned long long)(const void*)(K  + (bh * S_ + kb + 64) * HD_);
            const unsigned long long vbase =
                (unsigned long long)(const void*)(Vt + bh * HD_ * S_ + kb + 64);
            const unsigned klds = lds_addr(&Ktile[buf ^ 1][ldrow * KSTR + ldcol]);
            const unsigned vlds = lds_addr(&Vtile[buf ^ 1][ldrow * KSTR + ldcol]);
            async_b128(klds,      kgl,      kbase);
            async_b128(klds + 16, kgl + 16, kbase);
            async_b128(vlds,      vgl,      vbase);
            async_b128(vlds + 16, vgl + 16, vbase);
        }

        // mask values for this 64-key block (issued before the WMMA sequence)
        float mk[4];
        #pragma unroll
        for (int t = 0; t < 4; ++t) mk[t] = mask[b * S_ + kb + t * 16 + lo];

        const uint4* kt4 = reinterpret_cast<const uint4*>(Ktile[buf]);
        const uint4* vt4 = reinterpret_cast<const uint4*>(Vtile[buf]);

        // ---- S = (Q/gamma) K^T + mask : 4 tiles of 16 keys (K from LDS)
        v8f st[4];
        #pragma unroll
        for (int t = 0; t < 4; ++t) {
            const int r9 = (t * 16 + lo) * (KSTR / 8);     // KSTR=72 -> 9 uint4/row
            v16bf kb0 = frag_from(kt4[r9 + hi],     kt4[r9 + 2 + hi]);      // d 0..31
            v16bf kb1 = frag_from(kt4[r9 + 4 + hi], kt4[r9 + 6 + hi]);      // d 32..63
            v8f s = {};
            s = wmma_bf16(qa0, kb0, s);
            s = wmma_bf16(qa1, kb1, s);
            #pragma unroll
            for (int r = 0; r < 8; ++r) s[r] += mk[t];
            st[t] = s;
        }

        // ---- online softmax: row max over 64 keys (4 tiles + 16-lane butterfly)
        #pragma unroll
        for (int r = 0; r < 8; ++r) {
            float mx = fmaxf(fmaxf(st[0][r], st[1][r]), fmaxf(st[2][r], st[3][r]));
            #pragma unroll
            for (int off = 8; off >= 1; off >>= 1)
                mx = fmaxf(mx, __shfl_xor(mx, off, 16));
            const float mn = fmaxf(m_run[r], mx);
            const float sc = __expf(m_run[r] - mn);
            m_run[r] = mn;
            l_run[r] *= sc;
            o0[r] *= sc; o1[r] *= sc; o2[r] *= sc; o3[r] *= sc;
        }

        // ---- P = exp(S - m), stage to per-wave LDS (C layout -> row-major bf16)
        float ps[8];
        #pragma unroll
        for (int r = 0; r < 8; ++r) ps[r] = 0.f;
        #pragma unroll
        for (int t = 0; t < 4; ++t) {
            #pragma unroll
            for (int r = 0; r < 8; ++r) {
                const float p = __expf(st[t][r] - m_run[r]);
                ps[r] += p;
                Pbuf[wave][(hi * 8 + r) * KSTR + t * 16 + lo] = f2bf(p);
            }
        }
        #pragma unroll
        for (int r = 0; r < 8; ++r) {
            float v = ps[r];
            #pragma unroll
            for (int off = 8; off >= 1; off >>= 1)
                v += __shfl_xor(v, off, 16);
            l_run[r] += v;
        }

        // ---- O += P V : P A-fragments from LDS (transposed view), V from LDS
        const uint4* p4 = reinterpret_cast<const uint4*>(Pbuf[wave]);
        const int pr9 = lo * (KSTR / 8);
        v16bf pa0 = frag_from(p4[pr9 + hi],     p4[pr9 + 2 + hi]);   // keys 0..31
        v16bf pa1 = frag_from(p4[pr9 + 4 + hi], p4[pr9 + 6 + hi]);   // keys 32..63
        #pragma unroll
        for (int t = 0; t < 4; ++t) {
            const int r9 = (t * 16 + lo) * (KSTR / 8);
            v16bf vb0 = frag_from(vt4[r9 + hi],     vt4[r9 + 2 + hi]);      // keys 0..31
            v16bf vb1 = frag_from(vt4[r9 + 4 + hi], vt4[r9 + 6 + hi]);      // keys 32..63
            v8f o = (t == 0) ? o0 : (t == 1) ? o1 : (t == 2) ? o2 : o3;
            o = wmma_bf16(pa0, vb0, o);
            o = wmma_bf16(pa1, vb1, o);
            if (t == 0) o0 = o; else if (t == 1) o1 = o; else if (t == 2) o2 = o; else o3 = o;
        }

        // ---- drain this wave's async copies, then block-wide sync
        if (pf) wait_async();
        __syncthreads();
    }

    // ---- normalize + write ctx as [B,S,H] f32 (heads re-merged via indexing)
    #pragma unroll
    for (int r = 0; r < 8; ++r) {
        const float inv = 1.0f / l_run[r];
        const int row = qbase + hi * 8 + r;
        float* orow = out + ((size_t)b * S_ + row) * H_ + h * HD_;
        orow[ 0 + lo] = o0[r] * inv;
        orow[16 + lo] = o1[r] * inv;
        orow[32 + lo] = o2[r] * inv;
        orow[48 + lo] = o3[r] * inv;
    }
}

// ---------------------------------------------------------------- launcher
extern "C" void kernel_launch(void* const* d_in, const int* in_sizes, int n_in,
                              void* d_out, int out_size, void* d_ws, size_t ws_size,
                              hipStream_t stream) {
    const float* hidden = (const float*)d_in[0];
    const float* amask  = (const float*)d_in[1];
    const float* Wq     = (const float*)d_in[2];
    const float* bq     = (const float*)d_in[3];
    const float* Wk     = (const float*)d_in[4];
    const float* bk     = (const float*)d_in[5];
    const float* Wv     = (const float*)d_in[6];
    const float* bv     = (const float*)d_in[7];
    float* out = (float*)d_out;

    // workspace layout (bf16): X | W[3] | Q | K | V^T   (~54 MB total)
    us* Xbf = (us*)d_ws;
    us* Wbf = Xbf + (size_t)MT_ * H_;
    us* Qbf = Wbf + (size_t)3 * H_ * H_;
    us* Kbf = Qbf + (size_t)B_ * NH_ * S_ * HD_;
    us* Vtb = Kbf + (size_t)B_ * NH_ * S_ * HD_;

    // stage 1: f32 -> bf16
    const int nX = MT_ * H_;        // 6291456
    const int nW = H_ * H_;         // 589824
    cvt_bf16<<<nX / (256 * 8), 256, 0, stream>>>(hidden, Xbf, nX);
    cvt_bf16<<<nW / (256 * 8), 256, 0, stream>>>(Wq, Wbf,          nW);
    cvt_bf16<<<nW / (256 * 8), 256, 0, stream>>>(Wk, Wbf + nW,     nW);
    cvt_bf16<<<nW / (256 * 8), 256, 0, stream>>>(Wv, Wbf + 2 * nW, nW);

    // stage 2: QKV projections (128*48*3 wave tiles / 8 waves per block)
    qkv_gemm<<<(128 * 48 * 3) / 8, 256, 0, stream>>>(Xbf, Wbf, bq, bk, bv, Qbf, Kbf, Vtb);

    // stage 3: flash attention (one block per (b,h,128-row q block))
    flash_attn<<<B_ * NH_ * (S_ / 128), 256, 0, stream>>>(Qbf, Kbf, Vtb, amask, out);
}
